// Attention_1125281431646
// MI455X (gfx1250) — matrix operand order repaired and checked by reading, not compiled, not run on hardware
//
#include <hip/hip_runtime.h>

typedef __attribute__((ext_vector_type(16))) __bf16 v16bf;
typedef __attribute__((ext_vector_type(8)))  float  v8f;

#define BATCH 64
#define NQ    1024        // tokens (16 x 64 grid)
#define DIMK  512
#define WQKV_COLS 1536
#define HEADS 8
#define HD    64
#define QSCALE 0.125f     // HEAD_DIM^-0.5

// fp32 pair -> packed bf16 (RNE). Prefer native V_CVT_PK_BF16_F32.
__device__ __forceinline__ unsigned pack2(float lo, float hi) {
#if __has_builtin(__builtin_amdgcn_cvt_pk_bf16_f32)
  auto r = __builtin_amdgcn_cvt_pk_bf16_f32(lo, hi);
  unsigned u;
  __builtin_memcpy(&u, &r, sizeof(u));
  return u;
#else
  union { float f; unsigned u; } a, b;
  a.f = lo; b.f = hi;
  unsigned ra = (a.u + 0x7fffu + ((a.u >> 16) & 1u)) >> 16;
  unsigned rb = (b.u + 0x7fffu + ((b.u >> 16) & 1u)) >> 16;
  return (ra & 0xffffu) | (rb << 16);
#endif
}
__device__ __forceinline__ unsigned short bf1(float v) {
  return (unsigned short)(pack2(v, 0.f) & 0xffffu);
}

__device__ __forceinline__ void sched_fence() {
#if __has_builtin(__builtin_amdgcn_sched_barrier)
  __builtin_amdgcn_sched_barrier(0);
#endif
}

union ABu { unsigned u[8]; uint4 q4[2]; v16bf v; };

#define WAIT_DS() asm volatile("s_wait_dscnt 0x0" ::: "memory")

// ---------------------------------------------------------------------------
// Kernel 1: qkv = x @ w_qkv (bf16 WMMA, f32 accum). x and w tiles staged in
// LDS as bf16; epilogue staged through LDS -> b128 stores of Q (scaled),
// K row-major per head, V transposed per head.
// ---------------------------------------------------------------------------
__global__ __launch_bounds__(128)
void qkv_kernel(const float* __restrict__ x, const float* __restrict__ w,
                unsigned short* __restrict__ qws, unsigned short* __restrict__ kws,
                unsigned short* __restrict__ vtws) {
  __shared__ __align__(16) unsigned wTu[64 * 16];          // [col][kpair]
  __shared__ __align__(16) unsigned xlu[64 * 16];          // [row][kpair]
  __shared__ __align__(16) unsigned short stg[4][16 * 64]; // per-wave C tile
  const int wg   = blockIdx.x;
  const int ct   = wg % 24;
  const int ng   = (wg / 24) % 16;
  const int b    = wg / (24 * 16);
  const int col0 = ct * 64;
  const int wave = threadIdx.x >> 5;
  const int lane = threadIdx.x & 31;
  const int ntile = ng * 4 + wave;
  const int m     = lane & 15;
  const int kb2A  = (lane < 16) ? 0 : 8;   // A-layout K pair base (kbase/2)
  const int kb2B  = (lane < 16) ? 0 : 8;   // B-layout K pair base (kbB/2)

  // strength-reduced cooperative-load bases (per-thread constants)
  const int cc_w  = threadIdx.x & 63;      // w column within tile
  const int kk2w0 = threadIdx.x >> 6;      // 0..1
  const int kk2x  = threadIdx.x & 15;      // x k-pair index
  const int xrow0 = threadIdx.x >> 4;      // 0..7
  const float* wbase = w + (size_t)(2 * kk2w0) * WQKV_COLS + col0 + cc_w;
  const float* xbase = x + (size_t)(b * NQ + ng * 64 + xrow0) * DIMK + 2 * kk2x;
  unsigned* wTdst = &wTu[cc_w * 16 + kk2w0];
  unsigned* xldst = &xlu[xrow0 * 16 + kk2x];

  v8f acc[4] = {v8f{}, v8f{}, v8f{}, v8f{}};

  for (int k0 = 0; k0 < DIMK; k0 += 32) {
    __syncthreads();
    const float* wp = wbase + (size_t)k0 * WQKV_COLS;
    const float* xp = xbase + k0;
    #pragma unroll
    for (int i = 0; i < 8; ++i) {
      wTdst[2 * i] = pack2(wp[(size_t)(4 * i) * WQKV_COLS],
                           wp[(size_t)(4 * i + 1) * WQKV_COLS]);
      const float2 xv = *(const float2*)&xp[(size_t)(8 * i) * DIMK];
      xldst[128 * i] = pack2(xv.x, xv.y);
    }
    if (k0 + 32 < DIMK && threadIdx.x < 64) {  // prefetch next w tile
      int row = threadIdx.x >> 1, half = threadIdx.x & 1;
      __builtin_prefetch(&w[(size_t)(k0 + 32 + row) * WQKV_COLS + col0 + half * 32], 0, 1);
    }
    __syncthreads();

    ABu A;
    {
      const uint4* ap = (const uint4*)&xlu[(wave * 16 + m) * 16 + kb2A];
      A.q4[0] = ap[0]; A.q4[1] = ap[1];
    }
    ABu Bm[4];
    #pragma unroll
    for (int op = 0; op < 4; ++op) {
      const uint4* bp = (const uint4*)&wTu[(op * 16 + m) * 16 + kb2B];
      Bm[op].q4[0] = bp[0]; Bm[op].q4[1] = bp[1];
    }
    sched_fence();   // batch all ds_loads, one wait, then 4 wmmas
    #pragma unroll
    for (int op = 0; op < 4; ++op)
      acc[op] = __builtin_amdgcn_wmma_f32_16x16x32_bf16(
          false, A.v, false, Bm[op].v, (short)0, acc[op], false, false);
    sched_fence();
  }

  // epilogue: stage through LDS, then b128 stores
  const int type = col0 / 512;                // 0=q 1=k 2=v
  const int head = (col0 % 512) / 64;
  const int hi   = (lane < 16) ? 0 : 8;
  unsigned short* S = stg[wave];
  if (type < 2) {
    const float sc = (type == 0) ? QSCALE : 1.0f;
    #pragma unroll
    for (int op = 0; op < 4; ++op)
      #pragma unroll
      for (int r = 0; r < 8; ++r)
        S[(r + hi) * 64 + op * 16 + m] = bf1(acc[op][r] * sc);
    WAIT_DS();
    unsigned short* dst = (type == 0 ? qws : kws) +
        (size_t)(b * HEADS + head) * NQ * HD + (size_t)(ntile * 16) * HD;
    const uint4* sp = (const uint4*)S;
    #pragma unroll
    for (int c = 0; c < 4; ++c) {
      int f = c * 32 + lane;
      ((uint4*)dst)[f] = sp[f];
    }
  } else {
    #pragma unroll
    for (int op = 0; op < 4; ++op)
      #pragma unroll
      for (int r = 0; r < 8; ++r)
        S[(op * 16 + m) * 16 + r + hi] = bf1(acc[op][r]);   // [d][n]
    WAIT_DS();
    unsigned short* dst = vtws + (size_t)(b * HEADS + head) * HD * NQ + ntile * 16;
    const uint4* sp = (const uint4*)S;
    #pragma unroll
    for (int c = 0; c < 4; ++c) {
      int f = c * 32 + lane;
      int d = f >> 1, part = f & 1;
      *(uint4*)&dst[(size_t)d * NQ + part * 8] = sp[f];
    }
  }
}

// ---------------------------------------------------------------------------
// Kernel 2: windowed flash attention. One wave per (b, head, 16-query tile).
// ---------------------------------------------------------------------------
__global__ __launch_bounds__(128)
void attn_kernel(const unsigned short* __restrict__ qws,
                 const unsigned short* __restrict__ kws,
                 const unsigned short* __restrict__ vtws,
                 unsigned short* __restrict__ ows) {
  __shared__ __align__(16) unsigned short Pl[4][16 * 32];   // per-wave P tile
  __shared__ __align__(16) unsigned short Ost[4][16 * 64];  // per-wave O tile
  const int wave = threadIdx.x >> 5;
  const int lane = threadIdx.x & 31;
  const int qt   = (blockIdx.x & 15) * 4 + wave;
  const int h    = (blockIdx.x >> 4) & 7;
  const int b    = blockIdx.x >> 7;
  const int m     = lane & 15;
  const int kbase = (lane < 16) ? 0 : 16;
  const int kbB   = (lane < 16) ? 0 : 16;
  const int hi    = (lane < 16) ? 0 : 8;
  const int w0    = (qt & 3) * 16;
  const int h_img = qt >> 2;
  const size_t hoff = (size_t)(b * HEADS + h) * NQ * HD;
  const size_t vtb  = (size_t)(b * HEADS + h) * HD * NQ;
  unsigned short* Pw = Pl[wave];
  const unsigned short* kbasep = kws + hoff + (size_t)m * HD + kbB;   // + nkey*HD
  const unsigned short* vbasep = vtws + vtb + (size_t)m * NQ + kbB;   // + op*16*NQ + key0

  ABu QA[2];
  #pragma unroll
  for (int c = 0; c < 2; ++c) {
    const uint4* p = (const uint4*)&qws[hoff + (size_t)(qt * 16 + m) * HD + c * 32 + kbase];
    QA[c].q4[0] = p[0]; QA[c].q4[1] = p[1];
  }

  v8f O[4] = {v8f{}, v8f{}, v8f{}, v8f{}};
  float rmax[8], rsum[8];
  #pragma unroll
  for (int r = 0; r < 8; ++r) { rmax[r] = -1e30f; rsum[r] = 0.f; }

  for (int di = -3; di <= 3; ++di) {
    int i = h_img + di;
    if (i < 0 || i >= 16) continue;                   // uniform per wave
    for (int jt = 0; jt <= 32; jt += 32) {
      if (jt > w0 + 20 || jt + 31 < w0 - 5) continue; // tile outside window
      const int key0 = i * 64 + jt;

      // S = Q K^T : hoist all 4 K operands, then 4 WMMAs
      const unsigned short* krow = kbasep + (size_t)key0 * HD;
      ABu KB[2][2];
      #pragma unroll
      for (int nh = 0; nh < 2; ++nh)
        #pragma unroll
        for (int c = 0; c < 2; ++c) {
          const uint4* p = (const uint4*)(krow + nh * 16 * HD + c * 32);
          KB[nh][c].q4[0] = p[0]; KB[nh][c].q4[1] = p[1];
        }
      sched_fence();
      v8f S[2] = {v8f{}, v8f{}};
      #pragma unroll
      for (int nh = 0; nh < 2; ++nh)
        #pragma unroll
        for (int c = 0; c < 2; ++c)
          S[nh] = __builtin_amdgcn_wmma_f32_16x16x32_bf16(
              false, QA[c].v, false, KB[nh][c].v, (short)0, S[nh], false, false);
      sched_fence();

      // mask + online softmax (butterfly within 16-lane half = one M row)
      #pragma unroll
      for (int r = 0; r < 8; ++r) {
        int mq = r + hi;
        int wq = w0 + mq;
        int j0 = jt + m, j1 = jt + 16 + m;
        bool in0 = (j0 - wq <= 5) && (wq - j0 <= 5);
        bool in1 = (j1 - wq <= 5) && (wq - j1 <= 5);
        float s0 = in0 ? S[0][r] : -1e30f;
        float s1 = in1 ? S[1][r] : -1e30f;
        float tm = fmaxf(s0, s1);
        #pragma unroll
        for (int d = 1; d < 16; d <<= 1) tm = fmaxf(tm, __shfl_xor(tm, d, 32));
        float nm = fmaxf(rmax[r], tm);
        float cf = __expf(rmax[r] - nm);
        float p0 = in0 ? __expf(s0 - nm) : 0.f;
        float p1 = in1 ? __expf(s1 - nm) : 0.f;
        float ps = p0 + p1;
        #pragma unroll
        for (int d = 1; d < 16; d <<= 1) ps += __shfl_xor(ps, d, 32);
        rsum[r] = rsum[r] * cf + ps;
        rmax[r] = nm;
        #pragma unroll
        for (int op = 0; op < 4; ++op) O[op][r] *= cf;
        Pw[mq * 32 + m]      = bf1(p0);
        Pw[mq * 32 + 16 + m] = bf1(p1);
      }
      WAIT_DS();

      ABu PA;
      {
        const uint4* p = (const uint4*)&Pw[m * 32 + kbase];
        PA.q4[0] = p[0]; PA.q4[1] = p[1];
      }
      const unsigned short* vrow = vbasep + key0;
      ABu VB[4];
      #pragma unroll
      for (int op = 0; op < 4; ++op) {
        const uint4* p = (const uint4*)(vrow + (size_t)(op * 16) * NQ);
        VB[op].q4[0] = p[0]; VB[op].q4[1] = p[1];
      }
      sched_fence();
      #pragma unroll
      for (int op = 0; op < 4; ++op)
        O[op] = __builtin_amdgcn_wmma_f32_16x16x32_bf16(
            false, PA.v, false, VB[op].v, (short)0, O[op], false, false);
      sched_fence();
    }
  }

  // epilogue: normalize, stage through LDS, b128 stores
  unsigned short* S = Ost[wave];
  #pragma unroll
  for (int op = 0; op < 4; ++op)
    #pragma unroll
    for (int r = 0; r < 8; ++r)
      S[(r + hi) * 64 + op * 16 + m] = bf1(O[op][r] / rsum[r]);
  WAIT_DS();
  const uint4* sp = (const uint4*)S;
  #pragma unroll
  for (int c = 0; c < 4; ++c) {
    int f = c * 32 + lane;
    int row = f >> 3, col8 = (f & 7) * 8;
    *(uint4*)&ows[(size_t)(b * NQ + qt * 16 + row) * DIMK + h * HD + col8] = sp[f];
  }
}

// ---------------------------------------------------------------------------
// Kernel 3: out = O @ w_proj + b_proj (fp32 output via LDS-staged b128 stores)
// ---------------------------------------------------------------------------
__global__ __launch_bounds__(128)
void proj_kernel(const unsigned short* __restrict__ ows, const float* __restrict__ w,
                 const float* __restrict__ bias, float* __restrict__ out) {
  __shared__ __align__(16) unsigned wTu[64 * 16];   // [col][kpair]
  __shared__ __align__(16) float stgf[4][16 * 64];  // per-wave fp32 C tile
  const int wg   = blockIdx.x;
  const int ct   = wg % 8;
  const int ng   = (wg / 8) % 16;
  const int b    = wg / 128;
  const int col0 = ct * 64;
  const int wave = threadIdx.x >> 5;
  const int lane = threadIdx.x & 31;
  const int ntile = ng * 4 + wave;
  const int m     = lane & 15;
  const int kbase = (lane < 16) ? 0 : 16;
  const int kb2B  = (lane < 16) ? 0 : 8;

  const int cc_w  = threadIdx.x & 63;
  const int kk2w0 = threadIdx.x >> 6;
  const float* wbase = w + (size_t)(2 * kk2w0) * DIMK + col0 + cc_w;
  unsigned* wTdst = &wTu[cc_w * 16 + kk2w0];

  v8f acc[4] = {v8f{}, v8f{}, v8f{}, v8f{}};

  for (int k0 = 0; k0 < DIMK; k0 += 32) {
    __syncthreads();
    const float* wp = wbase + (size_t)k0 * DIMK;
    #pragma unroll
    for (int i = 0; i < 8; ++i)
      wTdst[2 * i] = pack2(wp[(size_t)(4 * i) * DIMK],
                           wp[(size_t)(4 * i + 1) * DIMK]);
    if (k0 + 32 < DIMK && threadIdx.x < 64) {
      int row = threadIdx.x >> 1, half = threadIdx.x & 1;
      __builtin_prefetch(&w[(size_t)(k0 + 32 + row) * DIMK + col0 + half * 32], 0, 1);
    }
    __syncthreads();

    ABu A;
    {
      const uint4* p = (const uint4*)&ows[(size_t)(b * NQ + ntile * 16 + m) * DIMK + k0 + kbase];
      A.q4[0] = p[0]; A.q4[1] = p[1];
    }
    ABu Bm[4];
    #pragma unroll
    for (int op = 0; op < 4; ++op) {
      const uint4* bp = (const uint4*)&wTu[(op * 16 + m) * 16 + kb2B];
      Bm[op].q4[0] = bp[0]; Bm[op].q4[1] = bp[1];
    }
    sched_fence();
    #pragma unroll
    for (int op = 0; op < 4; ++op)
      acc[op] = __builtin_amdgcn_wmma_f32_16x16x32_bf16(
          false, A.v, false, Bm[op].v, (short)0, acc[op], false, false);
    sched_fence();
  }

  const int hi = (lane < 16) ? 0 : 8;
  float* S = stgf[wave];
  #pragma unroll
  for (int op = 0; op < 4; ++op) {
    float bb = bias[col0 + op * 16 + m];
    #pragma unroll
    for (int r = 0; r < 8; ++r)
      S[(r + hi) * 64 + op * 16 + m] = acc[op][r] + bb;
  }
  WAIT_DS();
  const uint4* sp = (const uint4*)S;
  float* dst = out + (size_t)(b * NQ + ntile * 16) * DIMK + col0;
  #pragma unroll
  for (int c = 0; c < 8; ++c) {
    int f = c * 32 + lane;
    int row = f >> 4, col4 = (f & 15) * 4;
    *(uint4*)&dst[(size_t)row * DIMK + col4] = sp[f];
  }
}

// ---------------------------------------------------------------------------
extern "C" void kernel_launch(void* const* d_in, const int* in_sizes, int n_in,
                              void* d_out, int out_size, void* d_ws, size_t ws_size,
                              hipStream_t stream) {
  const float* x      = (const float*)d_in[0];
  const float* w_qkv  = (const float*)d_in[1];
  const float* w_proj = (const float*)d_in[2];
  const float* b_proj = (const float*)d_in[3];
  float* out = (float*)d_out;

  const size_t per = (size_t)BATCH * HEADS * NQ * HD;   // 33,554,432 bf16 elems
  unsigned short* qws  = (unsigned short*)d_ws;
  unsigned short* kws  = qws + per;
  unsigned short* vtws = kws + per;
  unsigned short* ows  = vtws + per;                    // [B][N][512] bf16

  qkv_kernel<<<dim3(BATCH * 16 * 24), dim3(128), 0, stream>>>(x, w_qkv, qws, kws, vtws);
  attn_kernel<<<dim3(BATCH * HEADS * 16), dim3(128), 0, stream>>>(qws, kws, vtws, ows);
  proj_kernel<<<dim3(BATCH * 16 * 8), dim3(128), 0, stream>>>(ows, w_proj, b_proj, out);
}
